// DepthwiseXCorr_824633721403
// MI455X (gfx1250) — compile-verified, matrix operand order checked by reading, not run on hardware
//
#include <hip/hip_runtime.h>
#include <hip/hip_bf16.h>
#include <cstdint>

typedef __attribute__((ext_vector_type(16))) _Float16 v16h;
typedef __attribute__((ext_vector_type(8)))  _Float16 v8h;
typedef __attribute__((ext_vector_type(8)))  float    v8f;
typedef __attribute__((ext_vector_type(4))) unsigned int u32x4;
typedef __attribute__((ext_vector_type(8))) int          i32x8;
typedef __attribute__((ext_vector_type(4))) int          i32x4;

#define TK   32
#define BM   128
#define BN   128
#define LDSW 40   // padded LDS row stride in halves: 32 data + 8 pad (4 dwords)
                  // row base = 20 dwords * r -> 16 distinct bank groups mod 64
#define TILEB (BM * LDSW * 2)  // 10240 bytes per padded 128x32 f16 tile

// ---- Tensor Data Mover availability (device pass only) ---------------------
#if defined(__has_builtin)
# if __has_builtin(__builtin_amdgcn_tensor_load_to_lds)
#  define USE_TDM 1
# endif
#endif
#ifndef USE_TDM
# define USE_TDM 0
#endif

#if USE_TDM
// TDM load of a 2D tile (tile_k x tile_rows, f16) from a row-major matrix
// (row stride = stride_elems) into LDS at byte offset lds_off.
// valid_rows < tile_rows => OOB rows zero-fill (tensor_dim1 check).
// LDS padding: 4 dwords every 16 dwords (64B row) -> bank-conflict-free
// LDSW-stride destination tile.
__device__ __forceinline__ void tdm_load_2d(const void* gptr, unsigned lds_off,
                                            int tile_k, int tile_rows,
                                            int valid_rows, int stride_elems) {
  unsigned long long ga = (unsigned long long)(uintptr_t)gptr;
  u32x4 g0 = {};
  g0.x = 1u;                                               // count=1 (user D#)
  g0.y = lds_off;                                          // lds_addr bytes
  g0.z = (unsigned)ga;                                     // global_addr[31:0]
  g0.w = (unsigned)((ga >> 32) & 0x01FFFFFFull) | 0x80000000u;  // [56:32] | type=2
  i32x8 g1 = {};
  g1[0] = (1 << 16)        // data_size = 2 bytes
        | (1 << 20)        // pad_enable
        | (3 << 22)        // pad_interval code 3 = every 16 dwords (64B row)
        | (3 << 25);       // pad_amount  code 3 = 4 dwords (16B)
  g1[1] = (tile_k & 0xFFFF) << 16;                         // tensor_dim0 lo16
  g1[2] = (valid_rows & 0xFFFF) << 16;                     // dim0 hi=0 | tensor_dim1 lo16
  g1[3] = (tile_k & 0xFFFF) << 16;                         // dim1 hi=0 | tile_dim0
  g1[4] = tile_rows & 0xFFFF;                              // tile_dim1 (tile_dim2=0)
  g1[5] = stride_elems;                                    // tensor_dim0_stride lo32
  g1[6] = 0;
  g1[7] = 0;
  i32x4 z4 = {};
#if __clang_major__ >= 23
  i32x8 z8 = {};
  __builtin_amdgcn_tensor_load_to_lds(g0, g1, z4, z4, z8, 0);
#else
  __builtin_amdgcn_tensor_load_to_lds(g0, g1, z4, z4, 0);
#endif
}
#endif  // USE_TDM

// Load a 16x32 f16 WMMA fragment from a padded LDS tile (row stride LDSW).
// CDNA5 16-bit A/B^T layout: lanes 0-15: row=lane, K={0..7,16..23};
// lanes 16-31: row=lane-16, K={8..15,24..31} -> two 16B ds reads per lane.
__device__ __forceinline__ v16h ldfrag(const _Float16* __restrict__ tile, int lane) {
  const _Float16* p = tile + (lane & 15) * LDSW + ((lane >> 4) << 3);
  v8h lo = *(const v8h*)(p);
  v8h hi = *(const v8h*)(p + 16);
  v16h f;
#pragma unroll
  for (int i = 0; i < 8; ++i) { f[i] = lo[i]; f[i + 8] = hi[i]; }
  return f;
}

// One 128x128x32 MMA step: 8 waves, wave tile 32(M) x 64(N) = 2x4 fragments.
__device__ __forceinline__ void mma_step(const _Float16* As, const _Float16* Bs,
                                         int wm, int wn, int lane, v8f (&acc)[2][4]) {
  v16h af[2], bf[4];
#pragma unroll
  for (int mf = 0; mf < 2; ++mf) af[mf] = ldfrag(As + (wm + mf * 16) * LDSW, lane);
#pragma unroll
  for (int nf = 0; nf < 4; ++nf) bf[nf] = ldfrag(Bs + (wn + nf * 16) * LDSW, lane);
#pragma unroll
  for (int mf = 0; mf < 2; ++mf)
#pragma unroll
    for (int nf = 0; nf < 4; ++nf)
      acc[mf][nf] = __builtin_amdgcn_wmma_f32_16x16x32_f16(
          false, af[mf], false, bf[nf], (short)0, acc[mf][nf], false, false);
}

// ---------------------------------------------------------------------------
// Fold BN(eval) into conv weights: wf = w * g/sqrt(v+eps) (f16),
// bias = b - m*scale (f32).  One block per output channel.
// ---------------------------------------------------------------------------
__global__ void fold_bn_weights(const float* __restrict__ w, const float* __restrict__ g,
                                const float* __restrict__ bb, const float* __restrict__ mm,
                                const float* __restrict__ vv, _Float16* __restrict__ wf,
                                float* __restrict__ bias, int K) {
  const int o = blockIdx.x;
  const float scale = g[o] * rsqrtf(vv[o] + 1e-5f);
  if (threadIdx.x == 0) bias[o] = bb[o] - mm[o] * scale;
  for (int kk = threadIdx.x; kk < K; kk += blockDim.x)
    wf[(size_t)o * K + kk] = (_Float16)(w[(size_t)o * K + kk] * scale);
}

__global__ void cvt_f16(const float* __restrict__ w, _Float16* __restrict__ wf, int n) {
  int i = blockIdx.x * blockDim.x + threadIdx.x;
  if (i < n) wf[i] = (_Float16)w[i];
}

// ---------------------------------------------------------------------------
// Implicit-GEMM 3x3 VALID conv + bias + ReLU (f16 WMMA, f32 accumulate).
//   M = B*OH*OW, N = Cout, K = Cin*9.  Tile 128x128, K-step 32.
//   Fully software-pipelined, double-buffered, one barrier per k-step:
//     issue TDM B(k+1) + gather A(k+1) -> WMMA(k) -> store A(k+1) ->
//     wait tensorcnt -> barrier.
// Dynamic LDS: As0|Bs0|As1|Bs1 (4 padded tiles) | koff[K*4B].
// ---------------------------------------------------------------------------
__global__ __launch_bounds__(256) void conv3x3_wmma(
    const _Float16* __restrict__ in, const _Float16* __restrict__ wf,
    const float* __restrict__ bias, _Float16* __restrict__ out,
    int B, int Cin, int IH, int IW, int OH, int OW, int Cout) {
  extern __shared__ char csm[];
  // NOTE: no constant-initialized pointer arrays here (ld.lld cannot relocate
  // static initializers containing LDS addrspacecasts) -- compute per use.
  auto asBuf = [&](int b) -> _Float16* { return (_Float16*)(csm + b * 2 * TILEB); };
  auto bsBuf = [&](int b) -> _Float16* { return (_Float16*)(csm + b * 2 * TILEB + TILEB); };
  int* koff = (int*)(csm + 4 * TILEB);

  const int tid  = threadIdx.x;
  const int lane = tid & 31;
  const int wave = tid >> 5;
  const int wm   = (wave & 3) * 32;
  const int wn   = (wave >> 2) * 64;

  const int M      = B * OH * OW;
  const int K      = Cin * 9;
  const int mbase  = blockIdx.x * BM;
  const int nbase  = blockIdx.y * BN;
  const int planeI = IH * IW;
  const int planeO = OH * OW;

  // Build k -> input-offset LUT once (c*planeI + kh*IW + kw).
  for (int k = tid; k < K; k += 256) {
    int c = k / 9, rem = k - c * 9, kh = rem / 3, kw = rem - kh * 3;
    koff[k] = c * planeI + kh * IW + kw;
  }
  __syncthreads();

  // Per-thread A assignment: 2 threads per tile row, 16 contiguous k each.
  const int arow  = tid >> 1;
  const int ahalf = (tid & 1) * 16;
  int am = mbase + arow;
  if (am >= M) am = M - 1;  // clamp: loads always valid, stores guarded
  const int ab  = am / planeO;
  const int ap  = am - ab * planeO;
  const int aoy = ap / OW;
  const int aox = ap - aoy * OW;
  const int aro = (ab * Cin * IH + aoy) * IW + aox;

  auto gatherA = [&](int k0, _Float16 (&v)[16]) {
    int ko[16];
#pragma unroll
    for (int q = 0; q < 4; ++q) {  // vector LUT reads -> ds_load_b128
      i32x4 t = *(const i32x4*)(koff + k0 + ahalf + 4 * q);
      ko[4 * q + 0] = t.x; ko[4 * q + 1] = t.y;
      ko[4 * q + 2] = t.z; ko[4 * q + 3] = t.w;
    }
#pragma unroll
    for (int j = 0; j < 16; ++j) v[j] = in[aro + ko[j]];
  };
  auto storeA = [&](_Float16* As, _Float16 (&v)[16]) {
    v8h h0, h1;
#pragma unroll
    for (int j = 0; j < 8; ++j) { h0[j] = v[j]; h1[j] = v[8 + j]; }
    *(v8h*)(As + arow * LDSW + ahalf)     = h0;
    *(v8h*)(As + arow * LDSW + ahalf + 8) = h1;
  };
#if !USE_TDM
  v8h bvals[2];
  auto loadB = [&](int k0) {
    const _Float16* src = wf + (size_t)(nbase + arow) * K + k0 + ahalf;
    bvals[0] = *(const v8h*)src;
    bvals[1] = *(const v8h*)(src + 8);
  };
  auto storeB = [&](_Float16* Bs) {
    *(v8h*)(Bs + arow * LDSW + ahalf)     = bvals[0];
    *(v8h*)(Bs + arow * LDSW + ahalf + 8) = bvals[1];
  };
#endif

  v8f acc[2][4] = {};
  const int ksteps = K / TK;  // 2304/32 = 72
  _Float16 vals[16];

  // Prologue: fill buffer 0.
  gatherA(0, vals);
#if USE_TDM
  if (wave == 0) tdm_load_2d(wf + (size_t)nbase * K, (unsigned)TILEB, TK, BN, BN, K);
#else
  loadB(0);
  storeB(bsBuf(0));
#endif
  storeA(asBuf(0), vals);
#if USE_TDM
  if (wave == 0) __builtin_amdgcn_s_wait_tensorcnt(0);
#endif
  __syncthreads();

  for (int ks = 0; ks < ksteps; ++ks) {
    const int buf = ks & 1, nbuf = buf ^ 1;
    const bool more = (ks + 1 < ksteps);
    if (more) {  // issue next tile's DMA + gather before this step's WMMAs
#if USE_TDM
      if (wave == 0)
        tdm_load_2d(wf + (size_t)nbase * K + (ks + 1) * TK,
                    (unsigned)(TILEB + nbuf * 2 * TILEB), TK, BN, BN, K);
#else
      loadB((ks + 1) * TK);
#endif
      gatherA((ks + 1) * TK, vals);
    }
    mma_step(asBuf(buf), bsBuf(buf), wm, wn, lane, acc);  // loads retire underneath
    if (more) {
      storeA(asBuf(nbuf), vals);
#if !USE_TDM
      storeB(bsBuf(nbuf));
#endif
    }
#if USE_TDM
    if (wave == 0) __builtin_amdgcn_s_wait_tensorcnt(0);  // DMA landed during WMMAs
#endif
    __syncthreads();  // single barrier per k-step
  }

  // Epilogue: bias + ReLU, scatter f16 NCHW.
  const int mrow = (lane >> 4) * 8;
  const int ncol = lane & 15;
#pragma unroll
  for (int mf = 0; mf < 2; ++mf) {
#pragma unroll
    for (int nf = 0; nf < 4; ++nf) {
      int n = nbase + wn + nf * 16 + ncol;
      float bv = bias[n];
#pragma unroll
      for (int r = 0; r < 8; ++r) {
        int m = mbase + wm + mf * 16 + mrow + r;
        if (m < M) {
          float x = acc[mf][nf][r] + bv;
          x = x > 0.f ? x : 0.f;
          int b  = m / planeO;
          int p  = m - b * planeO;
          int oy = p / OW;
          int ox = p - oy * OW;
          out[(((size_t)b * Cout + n) * OH + oy) * OW + ox] = (_Float16)x;
        }
      }
    }
  }
}

// ---------------------------------------------------------------------------
// Depthwise 5x5 cross-correlation, one block per (b,c) plane.
// s: (B,C,29,29) f16, k: (B,C,5,5) f16 -> channel-last (B*625, C) f16.
// ---------------------------------------------------------------------------
__global__ __launch_bounds__(256) void xcorr_dw(
    const _Float16* __restrict__ s, const _Float16* __restrict__ k,
    _Float16* __restrict__ out, int C) {
  __shared__ _Float16 sP[29 * 29];
  __shared__ float    kP[25];
  const int plane = blockIdx.x;  // b*C + c
  const int b = plane / C;
  const int c = plane - b * C;
  const _Float16* sp = s + (size_t)plane * 841;
  const _Float16* kp = k + (size_t)plane * 25;
  for (int e = threadIdx.x; e < 841; e += 256) sP[e] = sp[e];
  if (threadIdx.x < 25) kP[threadIdx.x] = (float)kp[threadIdx.x];
  __syncthreads();
  for (int m = threadIdx.x; m < 625; m += 256) {
    int y = m / 25, x = m - y * 25;
    float acc = 0.f;
#pragma unroll
    for (int i = 0; i < 5; ++i)
#pragma unroll
      for (int j = 0; j < 5; ++j)
        acc += (float)sP[(y + i) * 29 + (x + j)] * kP[i * 5 + j];
    out[((size_t)(b * 625 + m)) * C + c] = (_Float16)acc;
  }
}

// ---------------------------------------------------------------------------
// Head GEMM (1x1 conv): C[M,N] = A[M,K] * Bw[N,K]^T (+bias, opt ReLU).
//   Double-buffered TDM pipeline, one barrier per k-step:
//     issue DMA(k+1) -> WMMA(k) -> wait tensorcnt -> barrier.
// mode A (outH != 0): relu(acc+bias) -> f16 channel-last [m*N+n]
// mode B (outF != 0): acc+bias -> f32 NCHW with m = b*spatial + p
// ---------------------------------------------------------------------------
__global__ __launch_bounds__(256) void gemm_head(
    const _Float16* __restrict__ A, const _Float16* __restrict__ Bw,
    const float* __restrict__ bias, _Float16* __restrict__ outH,
    float* __restrict__ outF, int M, int N, int K, int spatial) {
  extern __shared__ char gsm[];

  const int tid  = threadIdx.x;
  const int lane = tid & 31;
  const int wave = tid >> 5;
  const int wm   = (wave & 3) * 32;
  const int wn   = (wave >> 2) * 64;
  const int mbase = blockIdx.x * BM;
  const int nbase = blockIdx.y * BN;
  const int validM = (M - mbase < BM) ? (M - mbase) : BM;
  const int arow  = tid >> 1;
  const int ahalf = (tid & 1) * 16;
  (void)arow; (void)ahalf;

  v8f acc[2][4] = {};
  const int ksteps = K / TK;  // 256/32 = 8

#if USE_TDM
  if (wave == 0) {
    tdm_load_2d(A + (size_t)mbase * K,  0u,              TK, BM, validM, K);
    tdm_load_2d(Bw + (size_t)nbase * K, (unsigned)TILEB, TK, BN, BN,     K);
    __builtin_amdgcn_s_wait_tensorcnt(0);
  }
  __syncthreads();
  for (int ks = 0; ks < ksteps; ++ks) {
    const int buf = ks & 1, nbuf = buf ^ 1;
    if (ks + 1 < ksteps && wave == 0) {  // DMA next slice under this step's WMMAs
      const int k1 = (ks + 1) * TK;
      tdm_load_2d(A + (size_t)mbase * K + k1,
                  (unsigned)(nbuf * 2 * TILEB), TK, BM, validM, K);
      tdm_load_2d(Bw + (size_t)nbase * K + k1,
                  (unsigned)(nbuf * 2 * TILEB + TILEB), TK, BN, BN, K);
    }
    const _Float16* As = (const _Float16*)(gsm + buf * 2 * TILEB);
    const _Float16* Bs = (const _Float16*)(gsm + buf * 2 * TILEB + TILEB);
    mma_step(As, Bs, wm, wn, lane, acc);
    if (wave == 0) __builtin_amdgcn_s_wait_tensorcnt(0);
    __syncthreads();  // single barrier per k-step
  }
#else
  v8h av[2], bv2[2];
  auto loadAB = [&](int k0) {
    int m = mbase + arow;
    av[0] = v8h{}; av[1] = v8h{};
    if (m < M) {
      const _Float16* src = A + (size_t)m * K + k0 + ahalf;
      av[0] = *(const v8h*)src;
      av[1] = *(const v8h*)(src + 8);
    }
    const _Float16* srcB = Bw + (size_t)(nbase + arow) * K + k0 + ahalf;
    bv2[0] = *(const v8h*)srcB;
    bv2[1] = *(const v8h*)(srcB + 8);
  };
  auto storeAB = [&](int b) {
    _Float16* As = (_Float16*)(gsm + b * 2 * TILEB);
    _Float16* Bs = As + TILEB / 2;
    *(v8h*)(As + arow * LDSW + ahalf)     = av[0];
    *(v8h*)(As + arow * LDSW + ahalf + 8) = av[1];
    *(v8h*)(Bs + arow * LDSW + ahalf)     = bv2[0];
    *(v8h*)(Bs + arow * LDSW + ahalf + 8) = bv2[1];
  };
  loadAB(0);
  storeAB(0);
  __syncthreads();
  for (int ks = 0; ks < ksteps; ++ks) {
    const int buf = ks & 1, nbuf = buf ^ 1;
    if (ks + 1 < ksteps) loadAB((ks + 1) * TK);
    const _Float16* As = (const _Float16*)(gsm + buf * 2 * TILEB);
    mma_step(As, As + TILEB / 2, wm, wn, lane, acc);
    if (ks + 1 < ksteps) storeAB(nbuf);
    __syncthreads();
  }
#endif

  const int mrow = (lane >> 4) * 8;
  const int ncol = lane & 15;
#pragma unroll
  for (int mf = 0; mf < 2; ++mf) {
#pragma unroll
    for (int nf = 0; nf < 4; ++nf) {
      int n = nbase + wn + nf * 16 + ncol;
      float bv = bias[n];
#pragma unroll
      for (int r = 0; r < 8; ++r) {
        int m = mbase + wm + mf * 16 + mrow + r;
        if (m < M) {
          float x = acc[mf][nf][r] + bv;
          if (outH) {
            x = x > 0.f ? x : 0.f;
            outH[(size_t)m * N + n] = (_Float16)x;
          } else {
            int b = m / spatial;
            int p = m - b * spatial;
            outF[((size_t)b * N + n) * spatial + p] = x;
          }
        }
      }
    }
  }
}

// ---------------------------------------------------------------------------
extern "C" void kernel_launch(void* const* d_in, const int* in_sizes, int n_in,
                              void* d_out, int out_size, void* d_ws, size_t ws_size,
                              hipStream_t stream) {
  const float* x_kernel = (const float*)d_in[0];
  const float* x_search = (const float*)d_in[1];
  const float* wk = (const float*)d_in[2];
  const float* gk = (const float*)d_in[3];
  const float* bk = (const float*)d_in[4];
  const float* mk = (const float*)d_in[5];
  const float* vk = (const float*)d_in[6];
  const float* wsw = (const float*)d_in[7];
  const float* gs = (const float*)d_in[8];
  const float* bs = (const float*)d_in[9];
  const float* ms = (const float*)d_in[10];
  const float* vs = (const float*)d_in[11];
  const float* w1 = (const float*)d_in[12];
  const float* g1 = (const float*)d_in[13];
  const float* b1 = (const float*)d_in[14];
  const float* m1 = (const float*)d_in[15];
  const float* v1 = (const float*)d_in[16];
  const float* w2 = (const float*)d_in[17];
  const float* b2 = (const float*)d_in[18];

  const int B = 64, Cin = 256, H = 256, Cout = 256;
  const int Kconv = Cin * 9;  // 2304
  const int Mk = B * 5 * 5;   // 1600
  const int Ms = B * 29 * 29; // 53824
  const int Mh = B * 25 * 25; // 40000
  const int nKer = B * Cin * 7 * 7;    // 802816
  const int nSrch = B * Cin * 31 * 31; // 15745024

  size_t off = 0;
  auto grab = [&](size_t bytes) -> void* {
    void* p = (char*)d_ws + off;
    off = (off + bytes + 255) & ~(size_t)255;
    return p;
  };
  _Float16* wkf   = (_Float16*)grab((size_t)H * Kconv * 2);
  _Float16* wsf   = (_Float16*)grab((size_t)H * Kconv * 2);
  _Float16* w1f   = (_Float16*)grab((size_t)H * H * 2);
  _Float16* w2f   = (_Float16*)grab((size_t)Cout * H * 2);
  float*    biask = (float*)grab(H * 4);
  float*    biass = (float*)grab(H * 4);
  float*    bias1 = (float*)grab(H * 4);
  _Float16* kerh  = (_Float16*)grab((size_t)nKer * 2);
  _Float16* srchh = (_Float16*)grab((size_t)nSrch * 2);
  _Float16* kfeat = (_Float16*)grab((size_t)B * H * 25 * 2);
  _Float16* sfeat = (_Float16*)grab((size_t)B * H * 841 * 2);
  _Float16* xfeat = (_Float16*)grab((size_t)Mh * H * 2);
  _Float16* feat2 = (_Float16*)grab((size_t)Mh * H * 2);
  (void)ws_size; (void)in_sizes; (void)n_in; (void)out_size;

  // 1) Fold BN into conv weights; convert weights + inputs to f16.
  fold_bn_weights<<<H, 256, 0, stream>>>(wk, gk, bk, mk, vk, wkf, biask, Kconv);
  fold_bn_weights<<<H, 256, 0, stream>>>(wsw, gs, bs, ms, vs, wsf, biass, Kconv);
  fold_bn_weights<<<H, 256, 0, stream>>>(w1, g1, b1, m1, v1, w1f, bias1, H);
  cvt_f16<<<(Cout * H + 255) / 256, 256, 0, stream>>>(w2, w2f, Cout * H);
  cvt_f16<<<(nKer + 255) / 256, 256, 0, stream>>>(x_kernel, kerh, nKer);
  cvt_f16<<<(nSrch + 255) / 256, 256, 0, stream>>>(x_search, srchh, nSrch);

  // 2) Twin conv branches (implicit GEMM + WMMA, pipelined TDM weights).
  const size_t convShm = 4 * TILEB + (size_t)Kconv * 4;  // 50176B
  {
    dim3 grid((Mk + BM - 1) / BM, Cout / BN);
    conv3x3_wmma<<<grid, 256, convShm, stream>>>(kerh, wkf, biask, kfeat,
                                                 B, Cin, 7, 7, 5, 5, H);
  }
  {
    dim3 grid((Ms + BM - 1) / BM, Cout / BN);
    conv3x3_wmma<<<grid, 256, convShm, stream>>>(srchh, wsf, biass, sfeat,
                                                 B, Cin, 31, 31, 29, 29, H);
  }

  // 3) Depthwise xcorr -> channel-last (Mh, H) f16.
  xcorr_dw<<<B * H, 256, 0, stream>>>(sfeat, kfeat, xfeat, H);

  // 4) Head: 1x1 conv + BN + ReLU (GEMM), then 1x1 conv + bias -> f32 NCHW.
  {
    dim3 grid((Mh + BM - 1) / BM, H / BN);
    gemm_head<<<grid, 256, 4 * TILEB, stream>>>(xfeat, w1f, bias1, feat2, nullptr,
                                                Mh, H, H, 0);
    gemm_head<<<grid, 256, 4 * TILEB, stream>>>(feat2, w2f, b2, nullptr, (float*)d_out,
                                                Mh, Cout, H, 625);
  }
}
